// SelfAttention2D_81037442941514
// MI455X (gfx1250) — compile-verified, matrix operand order checked by reading, not run on hardware
//
#include <hip/hip_runtime.h>

// ---------------------------------------------------------------------------
// SelfAttention2D for MI455X (gfx1250, wave32, WMMA).
// B=4, C=256, C8=32, N=64*64=4096.
//
//  * All three GEMM stages on v_wmma_f32_16x16x32_f16 (f16 in, f32 acc).
//  * Score matrix never touches HBM: each workgroup keeps a [16 x 4096] f32
//    row-block in LDS (~264KB of the 320KB/WGP), exact softmax in place,
//    Out-GEMM B-fragments read straight from LDS.
//  * Phase C: 4 independent accumulators per wave (breaks WMMA RAW chain,
//    4x reuse of each LDS B-fragment) -- critical at 1 workgroup/WGP.
//  * Softmax: 16 rows x 8 threads in parallel, 3 barriers total.
//  * x->xT transpose staged through an LDS tile for coalescing.
// ---------------------------------------------------------------------------

typedef __attribute__((ext_vector_type(16))) _Float16 v16h;
typedef __attribute__((ext_vector_type(8)))  _Float16 v8h;
typedef __attribute__((ext_vector_type(8)))  float    v8f;

#define B_DIM  4
#define C_DIM  256
#define C8_DIM 32
#define N_DIM  4096
#define NT     (N_DIM / 16)     // 256 column tiles
#define SE_STR (N_DIM + 8)      // padded LDS row stride (bank de-phasing)

// ---- fragment loaders -----------------------------------------------------
// 16-bit A/B fragment, wave32: lane holds row/col (lane&15); the 16 halves
// live as two contiguous 8-half chunks at K = 8*laneHi and K = 16 + 8*laneHi.

__device__ __forceinline__ v16h frag_from_h16(const _Float16* base, int laneHi) {
    v8h c0 = *(const v8h*)(base + 8 * laneHi);
    v8h c1 = *(const v8h*)(base + 16 + 8 * laneHi);
    v16h r;
#pragma unroll
    for (int e = 0; e < 8; ++e) { r[e] = c0[e]; r[e + 8] = c1[e]; }
    return r;
}

__device__ __forceinline__ v16h frag_from_f32(const float* base, int laneHi) {
    v16h r;
#pragma unroll
    for (int e = 0; e < 8; ++e) {
        r[e]     = (_Float16)base[8 * laneHi + e];
        r[e + 8] = (_Float16)base[16 + 8 * laneHi + e];
    }
    return r;
}

// ---- kernel 0: x [B,C,N] f32 -> xT [B,N,C] f16 via LDS tile transpose -----

__global__ void sa2d_pack_x(const float* __restrict__ x, _Float16* __restrict__ xT) {
    __shared__ _Float16 tile[64][66];                 // pad 2 halves: bank-clean
    int tilesPerB = (C_DIM / 64) * (N_DIM / 64);      // 4 * 64 = 256
    int b   = blockIdx.x / tilesPerB;
    int rem = blockIdx.x % tilesPerB;
    int c0  = (rem / (N_DIM / 64)) * 64;
    int n0  = (rem % (N_DIM / 64)) * 64;

    for (int e = threadIdx.x; e < 64 * 64; e += 256) {   // coalesced along n
        int cl = e >> 6, nl = e & 63;
        tile[nl][cl] = (_Float16)x[((long)(b * C_DIM + c0 + cl)) * N_DIM + n0 + nl];
    }
    __syncthreads();
    for (int e = threadIdx.x; e < 64 * 64; e += 256) {   // coalesced along c
        int nl = e >> 6, cl = e & 63;
        xT[((long)(b * N_DIM + n0 + nl)) * C_DIM + c0 + cl] = tile[nl][cl];
    }
}

// ---- kernel 1: q/k/v projections via WMMA ---------------------------------
// One block per (b, 16-col n-tile); 8 waves cover 20 output tiles
// (2 q, 2 k, 16 v). K = C = 256 -> 8 WMMA steps of 32.

__global__ void sa2d_project(const float* __restrict__ Wq, const float* __restrict__ bq,
                             const float* __restrict__ Wk, const float* __restrict__ bk,
                             const float* __restrict__ Wv, const float* __restrict__ bv,
                             const _Float16* __restrict__ xT,
                             _Float16* __restrict__ qT, _Float16* __restrict__ kT,
                             _Float16* __restrict__ vF) {
    int b  = blockIdx.x / NT;
    int nt = blockIdx.x % NT;
    int n0 = nt * 16;
    int wave   = threadIdx.x >> 5;
    int lane   = threadIdx.x & 31;
    int laneLo = lane & 15;
    int laneHi = lane >> 4;

    const _Float16* xrow = xT + ((long)(b * N_DIM + n0 + laneLo)) * C_DIM;

    for (int t = wave; t < 20; t += 8) {
        const float* W; const float* bias; int ot, kind;
        if (t < 2)      { W = Wq; bias = bq; ot = t;     kind = 0; }
        else if (t < 4) { W = Wk; bias = bk; ot = t - 2; kind = 1; }
        else            { W = Wv; bias = bv; ot = t - 4; kind = 2; }

        v8f acc = {};
        const float* wrow = W + (ot * 16 + laneLo) * C_DIM;
#pragma unroll
        for (int s = 0; s < C_DIM / 32; ++s) {
            v16h a  = frag_from_f32(wrow + s * 32, laneHi);
            v16h bm = frag_from_h16(xrow + s * 32, laneHi);
            acc = __builtin_amdgcn_wmma_f32_16x16x32_f16(
                false, a, false, bm, (short)0, acc, false, false);
        }
        int n = n0 + laneLo;
        if (kind < 2) {
            // o = ot*16 + 8*laneHi + r is contiguous -> one packed b128 store
            v8h pk;
#pragma unroll
            for (int r = 0; r < 8; ++r)
                pk[r] = (_Float16)(acc[r] + bias[ot * 16 + 8 * laneHi + r]);
            _Float16* dst = (kind == 0 ? qT : kT)
                          + ((long)(b * N_DIM + n)) * C8_DIM + ot * 16 + 8 * laneHi;
            *(v8h*)dst = pk;
        } else {
#pragma unroll
            for (int r = 0; r < 8; ++r) {
                int o = ot * 16 + 8 * laneHi + r;
                vF[((long)(b * C_DIM + o)) * N_DIM + n] =
                    (_Float16)(acc[r] + bias[o]);
            }
        }
    }
}

// ---- kernel 2: fused energy + softmax + Out GEMM --------------------------
// One block per (b, 16-row i-block). 4 waves, dynamic LDS:
//   sE  : [16][SE_STR] f32 score row-block (~263KB)
//   sRedA/sRedB : [128] f32 reduction scratch

__global__ void sa2d_attention(const _Float16* __restrict__ qT,
                               const _Float16* __restrict__ kT,
                               const _Float16* __restrict__ vF,
                               const float* __restrict__ x,
                               const float* __restrict__ gamma,
                               float* __restrict__ out,
                               float* __restrict__ attnMap) {
    extern __shared__ float smem[];
    float* sE    = smem;                     // 16 * SE_STR floats
    float* sRedA = smem + 16 * SE_STR;       // 128
    float* sRedB = sRedA + 128;              // 128

    int b  = blockIdx.x / NT;
    int it = blockIdx.x % NT;
    int i0 = it * 16;
    int tid    = threadIdx.x;
    int wave   = tid >> 5;
    int lane   = tid & 31;
    int laneLo = lane & 15;
    int laneHi = lane >> 4;

    // ---- phase A: energy row-block. E[i, j] = sum_c q[c,i] k[c,j]; K=C8=32.
    v16h aQ = frag_from_h16(qT + ((long)(b * N_DIM + i0 + laneLo)) * C8_DIM, laneHi);
    for (int jt = wave; jt < NT; jt += 4) {
        int j0 = jt * 16;
        v16h bK = frag_from_h16(kT + ((long)(b * N_DIM + j0 + laneLo)) * C8_DIM, laneHi);
        v8f acc = {};
        acc = __builtin_amdgcn_wmma_f32_16x16x32_f16(
            false, aQ, false, bK, (short)0, acc, false, false);
#pragma unroll
        for (int r = 0; r < 8; ++r)
            sE[(r + 8 * laneHi) * SE_STR + j0 + laneLo] = acc[r];
    }
    __syncthreads();

    // ---- phase B: exact softmax, all 16 rows in parallel (8 threads/row).
    {
        int row = tid >> 3;          // 0..15
        int sub = tid & 7;           // 0..7
        float* rowp = sE + row * SE_STR;

        float mx = -3.4e38f;
        for (int j = sub; j < N_DIM; j += 8) mx = fmaxf(mx, rowp[j]);
        sRedA[tid] = mx;
        __syncthreads();
#pragma unroll
        for (int u = 0; u < 8; ++u) mx = fmaxf(mx, sRedA[row * 8 + u]);

        float sm = 0.f;
        for (int j = sub; j < N_DIM; j += 8) {
            float e = __expf(rowp[j] - mx);
            rowp[j] = e;
            sm += e;
        }
        sRedB[tid] = sm;
        __syncthreads();
        float tot = 0.f;
#pragma unroll
        for (int u = 0; u < 8; ++u) tot += sRedB[row * 8 + u];
        float inv = 1.0f / tot;

        for (int j = sub; j < N_DIM; j += 8) {
            float a = rowp[j] * inv;
            rowp[j] = a;
            if (b == 0) attnMap[(long)(i0 + row) * N_DIM + j] = a;
        }
        __syncthreads();
    }

    // ---- phase C: Out[c, i] = sum_j V[c,j] * A[i,j].  K = 4096, 128 steps.
    // 4 independent accumulators per wave: breaks the WMMA RAW chain and
    // reuses each LDS B-fragment (f32->f16 cvt) 4x.
    float g = gamma[0];
    {
        int ct0 = wave * 4;                       // this wave's 4 c-tiles
        const _Float16* vrow[4];
#pragma unroll
        for (int u = 0; u < 4; ++u)
            vrow[u] = vF + ((long)(b * C_DIM + (ct0 + u) * 16 + laneLo)) * N_DIM;
        const float* arow = sE + laneLo * SE_STR;

        v8f acc[4] = {};
        for (int s = 0; s < N_DIM / 32; ++s) {
            if ((s & 3) == 0) {
#pragma unroll
                for (int u = 0; u < 4; ++u)
                    __builtin_prefetch(vrow[u] + s * 32 + 128, 0, 1);
            }
            v16h bA = frag_from_f32(arow + s * 32, laneHi);   // shared B-frag
#pragma unroll
            for (int u = 0; u < 4; ++u) {
                v16h aV = frag_from_h16(vrow[u] + s * 32, laneHi);
                acc[u] = __builtin_amdgcn_wmma_f32_16x16x32_f16(
                    false, aV, false, bA, (short)0, acc[u], false, false);
            }
        }
#pragma unroll
        for (int u = 0; u < 4; ++u) {
#pragma unroll
            for (int r = 0; r < 8; ++r) {
                int c = (ct0 + u) * 16 + 8 * laneHi + r;
                long idx = ((long)(b * C_DIM + c)) * N_DIM + (i0 + laneLo);
                out[idx] = g * acc[u][r] + x[idx];            // residual
            }
        }
    }
}

// ---------------------------------------------------------------------------

extern "C" void kernel_launch(void* const* d_in, const int* in_sizes, int n_in,
                              void* d_out, int out_size, void* d_ws, size_t ws_size,
                              hipStream_t stream) {
    const float* x     = (const float*)d_in[0];
    const float* Wq    = (const float*)d_in[1];
    const float* bq    = (const float*)d_in[2];
    const float* Wk    = (const float*)d_in[3];
    const float* bk    = (const float*)d_in[4];
    const float* Wv    = (const float*)d_in[5];
    const float* bv    = (const float*)d_in[6];
    const float* gamma = (const float*)d_in[7];

    float* out  = (float*)d_out;
    float* attn = out + (long)B_DIM * C_DIM * N_DIM;   // outputs concatenated

    // workspace (f16): xT [B,N,C] | qT [B,N,C8] | kT [B,N,C8] | v [B,C,N]
    _Float16* xT = (_Float16*)d_ws;
    _Float16* qT = xT + (long)B_DIM * N_DIM * C_DIM;
    _Float16* kT = qT + (long)B_DIM * N_DIM * C8_DIM;
    _Float16* vF = kT + (long)B_DIM * N_DIM * C8_DIM;

    // 0) pack + transpose x to f16 (LDS tiled)
    sa2d_pack_x<<<B_DIM * (C_DIM / 64) * (N_DIM / 64), 256, 0, stream>>>(x, xT);

    // 1) q/k/v projections
    sa2d_project<<<B_DIM * NT, 256, 0, stream>>>(Wq, bq, Wk, bk, Wv, bv,
                                                 xT, qT, kT, vF);

    // 2) fused attention (~264 KB dynamic LDS per workgroup)
    size_t shmem = (size_t)(16 * SE_STR + 256) * sizeof(float);
    (void)hipFuncSetAttribute((const void*)sa2d_attention,
                              hipFuncAttributeMaxDynamicSharedMemorySize,
                              (int)shmem);
    sa2d_attention<<<B_DIM * NT, 128, shmem, stream>>>(qT, kT, vF, x, gamma,
                                                       out, attn);
}
